// ARDecoder_63556926046556
// MI455X (gfx1250) — matrix operand order repaired and checked by reading, not compiled, not run on hardware
//
#include <hip/hip_runtime.h>
#include <hip/hip_bf16.h>
#include <math.h>

// ---------------------------------------------------------------------------
// CDNA5 (gfx1250) implementation of the AR block-decoder forward pass.
// All GEMMs and attention run on v_wmma_f32_16x16x32_bf16 (wave32 fragments),
// f32 accumulate. Flash-style attention avoids materializing S x S scores.
// ---------------------------------------------------------------------------

typedef __bf16 bf16_t;
typedef bf16_t v16bf __attribute__((ext_vector_type(16)));
typedef float  v8f   __attribute__((ext_vector_type(8)));

#define NEG_INF (-__builtin_inff())

// Problem constants (from the reference)
#define BP_ 2
#define L_ 32
#define N_ 16
#define O_ 64
#define D_ 1024
#define AD_ 1024
#define AAN_ 8
#define HEAD_ 128
#define W_ 256
#define FH_ 2048
#define M_ 48          // O - N
#define S_ 2048        // L * O

// ---- WMMA fragment index helpers (per CDNA5 ISA 16-bit layouts, wave32) ----
// A-matrix 16x32: lanes 0-15 hold K {0..7,16..23}, lanes 16-31 hold {8..15,24..31}
__device__ __forceinline__ int kOfA(int lane, int e) {
  return ((lane & 16) ? 8 : 0) + (e < 8 ? e : e + 8);
}
// B-matrix 32x16: lanes 0-15 hold K {0..7,16..23}, lanes 16-31 hold {8..15,24..31}
__device__ __forceinline__ int kOfB(int lane, int e) {
  return ((lane & 16) ? 8 : 0) + (e < 8 ? e : e + 8);
}

__device__ __forceinline__ v8f wmma_bf16(v16bf a, v16bf b, v8f c) {
  return __builtin_amdgcn_wmma_f32_16x16x32_bf16(false, a, false, b, (short)0, c,
                                                 false, false);
}

__device__ __forceinline__ v8f zero8() {
  v8f z;
#pragma unroll
  for (int i = 0; i < 8; i++) z[i] = 0.0f;
  return z;
}

// ---------------------------------------------------------------------------
// Generic GEMM:  C[m][n] (+)= sum_k A[m][k] * B[n][k]     (B row-major, ldb=K)
// Row mappings: row r -> base + (r/G)*GS + (off + r%G)*ld  (identity: G=rows,GS=0,off=0)
// Block tile 128x64, 8 waves, each wave 32x32 via 2x2 WMMA 16x16x32 bf16.
// ---------------------------------------------------------------------------
__global__ __launch_bounds__(256) void gemm_wmma(
    const float* __restrict__ A, int lda, int GA, long GSA, int offA,
    const float* __restrict__ B,
    float* __restrict__ C, int ldc, int GC, long GSC, int offC,
    int K, int accflag)
{
  __shared__ bf16_t As[128][34];
  __shared__ bf16_t Bs[64][34];
  const int tid  = threadIdx.x;
  const int lane = tid & 31;
  const int wid  = tid >> 5;
  const int bm = blockIdx.x * 128;
  const int bn = blockIdx.y * 64;
  const int wrow = (wid >> 1) * 32;
  const int wcol = (wid & 1) * 32;

  v8f acc[2][2];
#pragma unroll
  for (int i = 0; i < 2; i++)
#pragma unroll
    for (int j = 0; j < 2; j++) acc[i][j] = zero8();

  for (int k0 = 0; k0 < K; k0 += 32) {
    __syncthreads();
    { // stage A tile 128x32 (f32 -> bf16)
      int r  = tid >> 1;
      int cc = (tid & 1) * 16;
      int grow = bm + r;
      const float* ap = A + (long)(grow / GA) * GSA +
                        (long)(offA + (grow % GA)) * lda + k0 + cc;
      __builtin_prefetch(ap + 32, 0, 1);
#pragma unroll
      for (int j = 0; j < 16; j += 4) {
        float4 f = *(const float4*)(ap + j);
        As[r][cc + j + 0] = (bf16_t)f.x;
        As[r][cc + j + 1] = (bf16_t)f.y;
        As[r][cc + j + 2] = (bf16_t)f.z;
        As[r][cc + j + 3] = (bf16_t)f.w;
      }
    }
    if (tid < 128) { // stage B tile 64x32
      int r  = tid >> 1;
      int cc = (tid & 1) * 16;
      const float* bp = B + (long)(bn + r) * K + k0 + cc;
      __builtin_prefetch(bp + 32, 0, 1);
#pragma unroll
      for (int j = 0; j < 16; j += 4) {
        float4 f = *(const float4*)(bp + j);
        Bs[r][cc + j + 0] = (bf16_t)f.x;
        Bs[r][cc + j + 1] = (bf16_t)f.y;
        Bs[r][cc + j + 2] = (bf16_t)f.z;
        Bs[r][cc + j + 3] = (bf16_t)f.w;
      }
    }
    __syncthreads();

    v16bf af[2], bfr[2];
#pragma unroll
    for (int i = 0; i < 2; i++) {
      int r = wrow + i * 16 + (lane & 15);
#pragma unroll
      for (int e = 0; e < 16; e++) af[i][e] = As[r][kOfA(lane, e)];
    }
#pragma unroll
    for (int j = 0; j < 2; j++) {
      int r = wcol + j * 16 + (lane & 15);
#pragma unroll
      for (int e = 0; e < 16; e++) bfr[j][e] = Bs[r][kOfB(lane, e)];
    }
#pragma unroll
    for (int i = 0; i < 2; i++)
#pragma unroll
      for (int j = 0; j < 2; j++)
        acc[i][j] = wmma_bf16(af[i], bfr[j], acc[i][j]);
  }

  // epilogue: C/D layout -> lane holds col (lane&15), rows r+(lane>=16?8:0)
  const int halfsel = (lane & 16) ? 8 : 0;
  const int nlo = lane & 15;
#pragma unroll
  for (int i = 0; i < 2; i++)
#pragma unroll
    for (int j = 0; j < 2; j++)
#pragma unroll
      for (int r = 0; r < 8; r++) {
        int m = bm + wrow + i * 16 + r + halfsel;
        int n = bn + wcol + j * 16 + nlo;
        float* cp = C + (long)(m / GC) * GSC +
                    (long)(offC + (m % GC)) * ldc + n;
        float v = acc[i][j][r];
        if (accflag) v += *cp;
        *cp = v;
      }
}

// ---------------------------------------------------------------------------
// RMSNorm over D=1024, one wave per row, with source-row mapping.
// ---------------------------------------------------------------------------
__global__ __launch_bounds__(256) void rmsnorm_k(
    const float* __restrict__ X, int ldx, int GX, long GSX, int offX,
    const float* __restrict__ w, float* __restrict__ Y)
{
  int row  = blockIdx.x * 8 + (threadIdx.x >> 5);
  int lane = threadIdx.x & 31;
  const float* xp = X + (long)(row / GX) * GSX + (long)(offX + (row % GX)) * ldx;
  float v[32];
  float ss = 0.0f;
#pragma unroll
  for (int j = 0; j < 8; j++) {
    float4 f = *(const float4*)(xp + lane * 4 + j * 128);
    v[4 * j + 0] = f.x; v[4 * j + 1] = f.y; v[4 * j + 2] = f.z; v[4 * j + 3] = f.w;
    ss += f.x * f.x + f.y * f.y + f.z * f.z + f.w * f.w;
  }
#pragma unroll
  for (int mm = 1; mm < 32; mm <<= 1) ss += __shfl_xor(ss, mm, 32);
  float sc = rsqrtf(ss * (1.0f / 1024.0f) + 1e-5f);
  float* yp = Y + (long)row * 1024;
#pragma unroll
  for (int j = 0; j < 8; j++) {
    float4 wf = *(const float4*)(w + lane * 4 + j * 128);
    float4 o;
    o.x = v[4 * j + 0] * sc * wf.x;
    o.y = v[4 * j + 1] * sc * wf.y;
    o.z = v[4 * j + 2] * sc * wf.z;
    o.w = v[4 * j + 3] * sc * wf.w;
    *(float4*)(yp + lane * 4 + j * 128) = o;
  }
}

// ---------------------------------------------------------------------------
// l2norm + rotary for q and k vectors in the qkv buffer (in place).
// One wave per (b,s,h,{q|k}) vector of 128. Lane holds dims {l,l+32,l+64,l+96}.
// ---------------------------------------------------------------------------
__global__ __launch_bounds__(256) void qk_prep(float* __restrict__ qkv)
{
  int vid  = blockIdx.x * 8 + (threadIdx.x >> 5);
  int lane = threadIdx.x & 31;
  const int T = BP_ * S_ * AAN_;
  int isK = (vid >= T) ? 1 : 0;
  int u = isK ? vid - T : vid;
  int b = u / (S_ * AAN_);
  int rem = u % (S_ * AAN_);
  int s = rem / AAN_, h = rem % AAN_;
  float* p = qkv + (long)(b * S_ + s) * (3 * AD_) + (isK ? AD_ : 0) + h * HEAD_;

  float x1a = p[lane], x1b = p[lane + 32];
  float x2a = p[lane + 64], x2b = p[lane + 96];
  float ss = x1a * x1a + x1b * x1b + x2a * x2a + x2b * x2b;
#pragma unroll
  for (int mm = 1; mm < 32; mm <<= 1) ss += __shfl_xor(ss, mm, 32);
  float sc = 1.0f / fmaxf(sqrtf(ss), 1e-5f);
  x1a *= sc; x1b *= sc; x2a *= sc; x2b *= sc;

  const float LOG_THETA = 9.210340371976184f; // ln(10000)
  {
    float inv = expf(-((float)lane / 64.0f) * LOG_THETA);
    float f = (float)s * inv, c = cosf(f), sn = sinf(f);
    p[lane]      =  x1a * c + x2a * sn;
    p[lane + 64] = -x1a * sn + x2a * c;
  }
  {
    float inv = expf(-((float)(lane + 32) / 64.0f) * LOG_THETA);
    float f = (float)s * inv, c = cosf(f), sn = sinf(f);
    p[lane + 32] =  x1b * c + x2b * sn;
    p[lane + 96] = -x1b * sn + x2b * c;
  }
}

// ---------------------------------------------------------------------------
// Attention mask per the reference build_mask().
// ---------------------------------------------------------------------------
__device__ __forceinline__ bool attn_mask(const int* sdoc, int q, int k)
{
  int qb = q >> 6, qr = q & 63;
  int kb = k >> 6, kr = k & 63;
  bool qnew = qr >= M_, knew = kr >= M_;
  int dq  = sdoc[qb];
  int dk  = sdoc[kb];
  int dkp = sdoc[kb > 0 ? kb - 1 : 0];
  int posq = qr + qb * N_, posk = kr + kb * N_;
  bool nn = qnew && knew && (q >= k) && (posq < posk + W_) && (dq == dk);
  bool ao = ((qnew && (dq == dkp)) || !qnew) && !knew && (qb == kb);
  return nn || ao;
}

// ---------------------------------------------------------------------------
// Flash attention. Block = 128 threads (4 waves); wave w owns q rows
// [qbase, qbase+16). K/V staged 32x128 to LDS per step; online softmax with
// half-wave shuffles; P goes through a per-wave LDS tile to reach the
// A-fragment layout for the PV WMMAs. Output accumulators: 16x128 f32.
// ---------------------------------------------------------------------------
__global__ __launch_bounds__(128) void attn_flash(
    const float* __restrict__ qkv, const int* __restrict__ doc,
    float* __restrict__ out)
{
  __shared__ bf16_t Ks[32][HEAD_ + 4];
  __shared__ bf16_t Vs[32][HEAD_ + 4];
  __shared__ bf16_t Ps[4][16][36];
  __shared__ int sdoc[L_];

  const int b = blockIdx.z, h = blockIdx.y;
  const int tid = threadIdx.x, lane = tid & 31, w = tid >> 5;
  const int qbase = blockIdx.x * 64 + w * 16;
  const int SQ = 3 * AD_;

  if (tid < L_) sdoc[tid] = doc[b * L_ + tid];
  __syncthreads();

  const float* base = qkv + (long)(b * S_) * SQ;
  const int mrow = lane & 15;
  const int nlo  = lane & 15;
  const int halfsel = (lane & 16) ? 8 : 0;

  // Q fragments (4 K-chunks of 32 over head dim 128)
  v16bf qf[4];
  {
    const float* qrow = base + (long)(qbase + mrow) * SQ + h * HEAD_;
#pragma unroll
    for (int c = 0; c < 4; c++)
#pragma unroll
      for (int e = 0; e < 16; e++)
        qf[c][e] = (bf16_t)qrow[c * 32 + kOfA(lane, e)];
  }

  v8f o[8];
#pragma unroll
  for (int t = 0; t < 8; t++) o[t] = zero8();
  float rm[8], rl[8];
#pragma unroll
  for (int r = 0; r < 8; r++) { rm[r] = NEG_INF; rl[r] = 0.0f; }

  const float scale = 0.08838834764831845f; // 1/sqrt(128)

  for (int kv = 0; kv < S_; kv += 32) {
    __syncthreads();
    { // stage K and V: 32 rows x 128 dims, f32 -> bf16
      int r  = tid >> 2;
      int cc = (tid & 3) * 32;
      const float* kp = base + (long)(kv + r) * SQ + AD_ + h * HEAD_ + cc;
      const float* vp = base + (long)(kv + r) * SQ + 2 * AD_ + h * HEAD_ + cc;
#pragma unroll
      for (int j = 0; j < 32; j += 4) {
        float4 f = *(const float4*)(kp + j);
        Ks[r][cc + j + 0] = (bf16_t)f.x; Ks[r][cc + j + 1] = (bf16_t)f.y;
        Ks[r][cc + j + 2] = (bf16_t)f.z; Ks[r][cc + j + 3] = (bf16_t)f.w;
        float4 g = *(const float4*)(vp + j);
        Vs[r][cc + j + 0] = (bf16_t)g.x; Vs[r][cc + j + 1] = (bf16_t)g.y;
        Vs[r][cc + j + 2] = (bf16_t)g.z; Vs[r][cc + j + 3] = (bf16_t)g.w;
      }
    }
    __syncthreads();

    // S tiles: s0 = Q*K^T (kv..kv+15), s1 = (kv+16..kv+31)
    v8f s0 = zero8(), s1 = zero8();
#pragma unroll
    for (int c = 0; c < 4; c++) {
      v16bf kb0, kb1;
#pragma unroll
      for (int e = 0; e < 16; e++) {
        int kk = c * 32 + kOfB(lane, e);
        kb0[e] = Ks[nlo][kk];
        kb1[e] = Ks[16 + nlo][kk];
      }
      s0 = wmma_bf16(qf[c], kb0, s0);
      s1 = wmma_bf16(qf[c], kb1, s1);
    }

    // mask + online softmax
    const int k0i = kv + nlo, k1i = kv + 16 + nlo;
    float p0[8], p1[8], nm[8];
#pragma unroll
    for (int r = 0; r < 8; r++) {
      int q_idx = qbase + r + halfsel;
      float v0 = attn_mask(sdoc, q_idx, k0i) ? s0[r] * scale : NEG_INF;
      float v1 = attn_mask(sdoc, q_idx, k1i) ? s1[r] * scale : NEG_INF;
      p0[r] = v0; p1[r] = v1;
      float t = fmaxf(v0, v1);
#pragma unroll
      for (int mm = 1; mm < 16; mm <<= 1) t = fmaxf(t, __shfl_xor(t, mm, 16));
      nm[r] = fmaxf(rm[r], t);
    }
#pragma unroll
    for (int r = 0; r < 8; r++) {
      float corr;
      if (nm[r] == NEG_INF) {            // fully masked so far
        corr = 1.0f; p0[r] = 0.0f; p1[r] = 0.0f;
      } else {
        corr = expf(rm[r] - nm[r]);      // rm=-inf -> 0
        p0[r] = expf(p0[r] - nm[r]);
        p1[r] = expf(p1[r] - nm[r]);
      }
      float rs = p0[r] + p1[r];
#pragma unroll
      for (int mm = 1; mm < 16; mm <<= 1) rs += __shfl_xor(rs, mm, 16);
      rl[r] = rl[r] * corr + rs;
      rm[r] = nm[r];
#pragma unroll
      for (int t = 0; t < 8; t++) o[t][r] *= corr;
      Ps[w][r + halfsel][nlo]      = (bf16_t)p0[r];
      Ps[w][r + halfsel][16 + nlo] = (bf16_t)p1[r];
    }
    __syncthreads();

    // P fragment (A layout) and PV accumulation over the 8 head-dim tiles
    v16bf pa;
#pragma unroll
    for (int e = 0; e < 16; e++) pa[e] = Ps[w][mrow][kOfA(lane, e)];
#pragma unroll
    for (int t = 0; t < 8; t++) {
      v16bf vb;
#pragma unroll
      for (int e = 0; e < 16; e++) vb[e] = Vs[kOfB(lane, e)][t * 16 + nlo];
      o[t] = wmma_bf16(pa, vb, o[t]);
    }
  }

  // epilogue: normalize and write (b, s, h*128 + d)
#pragma unroll
  for (int r = 0; r < 8; r++) {
    float inv = (rl[r] > 0.0f) ? (1.0f / rl[r]) : 0.0f;
    long row = (long)(b * S_ + qbase + r + halfsel);
#pragma unroll
    for (int t = 0; t < 8; t++)
      out[row * AD_ + h * HEAD_ + t * 16 + nlo] = o[t][r] * inv;
  }
}

// ---------------------------------------------------------------------------
// SwiGLU: g[r,f] = silu(up[r,f]) * up[r, FH+f]
// ---------------------------------------------------------------------------
__global__ __launch_bounds__(256) void silu_gate(
    const float* __restrict__ up, float* __restrict__ g)
{
  long i = (long)blockIdx.x * 256 + threadIdx.x;
  long r = i / FH_, f = i % FH_;
  float a = up[r * (2 * FH_) + f];
  float bb = up[r * (2 * FH_) + FH_ + f];
  g[i] = (a / (1.0f + expf(-a))) * bb;
}

// ---------------------------------------------------------------------------
// y2 = rmsnorm( where(cond, pad_emb, roll(y,1)), new_norm_w ).  j==1 case.
// One wave per output row (b, l*N+n).
// ---------------------------------------------------------------------------
__global__ __launch_bounds__(256) void build_y2(
    const float* __restrict__ x, const unsigned char* __restrict__ begin,
    const float* __restrict__ pad_emb, const float* __restrict__ w,
    float* __restrict__ y2)
{
  int row  = blockIdx.x * 8 + (threadIdx.x >> 5);  // 0..BP*L*N-1
  int lane = threadIdx.x & 31;
  int b = row / (L_ * N_);
  int t = row % (L_ * N_);
  const float* src;
  if ((t & (N_ - 1)) == 0 && begin[b * L_ + (t >> 4)]) {
    src = pad_emb;
  } else {
    int tt = (t == 0) ? (L_ * N_ - 1) : t - 1;   // roll by 1 within batch
    src = x + (long)(b * S_ + (tt >> 4) * O_ + M_ + (tt & (N_ - 1))) * AD_;
  }
  float v[32];
  float ss = 0.0f;
#pragma unroll
  for (int j = 0; j < 8; j++) {
    float4 f = *(const float4*)(src + lane * 4 + j * 128);
    v[4 * j + 0] = f.x; v[4 * j + 1] = f.y; v[4 * j + 2] = f.z; v[4 * j + 3] = f.w;
    ss += f.x * f.x + f.y * f.y + f.z * f.z + f.w * f.w;
  }
#pragma unroll
  for (int mm = 1; mm < 32; mm <<= 1) ss += __shfl_xor(ss, mm, 32);
  float sc = rsqrtf(ss * (1.0f / 1024.0f) + 1e-5f);
  float* yp = y2 + (long)row * AD_;
#pragma unroll
  for (int j = 0; j < 8; j++) {
    float4 wf = *(const float4*)(w + lane * 4 + j * 128);
    float4 oo;
    oo.x = v[4 * j + 0] * sc * wf.x;
    oo.y = v[4 * j + 1] * sc * wf.y;
    oo.z = v[4 * j + 2] * sc * wf.z;
    oo.w = v[4 * j + 3] * sc * wf.w;
    *(float4*)(yp + lane * 4 + j * 128) = oo;
  }
}

// ---------------------------------------------------------------------------
static inline void launch_gemm(hipStream_t st,
                               const float* A, int lda, int GA, long GSA, int offA,
                               const float* B,
                               float* C, int ldc, int GC, long GSC, int offC,
                               int M, int N, int K, int acc)
{
  dim3 grid(M / 128, N / 64);
  gemm_wmma<<<grid, 256, 0, st>>>(A, lda, GA, GSA, offA, B, C, ldc, GC, GSC, offC,
                                  K, acc);
}

extern "C" void kernel_launch(void* const* d_in, const int* in_sizes, int n_in,
                              void* d_out, int out_size, void* d_ws, size_t ws_size,
                              hipStream_t stream)
{
  (void)in_sizes; (void)n_in; (void)out_size; (void)ws_size;
  const float* x_input     = (const float*)d_in[0];
  const float* x_ar        = (const float*)d_in[1];
  const int*   doc         = (const int*)d_in[2];
  const unsigned char* begin = (const unsigned char*)d_in[3];
  const float* old_norm_w  = (const float*)d_in[4];
  const float* new_norm_w  = (const float*)d_in[5];
  const float* w_old       = (const float*)d_in[6];
  const float* w_ar        = (const float*)d_in[7];
  const float* w_de        = (const float*)d_in[8];
  const float* w_ar1       = (const float*)d_in[9];
  const float* w_de1       = (const float*)d_in[10];
  const float* w_y1        = (const float*)d_in[11];
  const float* pad_emb     = (const float*)d_in[12];
  const float* w_agg       = (const float*)d_in[13];
  const float* attn_norm_w = (const float*)d_in[14];
  const float* ffn_norm_w  = (const float*)d_in[15];
  const float* w_qkv       = (const float*)d_in[16];
  const float* w_o         = (const float*)d_in[17];
  const float* w_up        = (const float*)d_in[18];
  const float* w_down      = (const float*)d_in[19];

  // workspace layout (floats); total ~= 39.8M floats (~152 MB)
  float* ws    = (float*)d_ws;
  float* xbuf  = ws;                       // BP*S*AD        = 4194304
  float* xoldn = xbuf  + 4194304;          // BP*L*M*D       = 3145728
  float* xarn  = xoldn + 3145728;          // BP*L*N*D       = 1048576
  float* xden  = xarn  + 1048576;          // BP*L*N*D       = 1048576
  float* hbuf  = xden  + 1048576;          // BP*S*AD        = 4194304
  float* y2buf = hbuf  + 4194304;          // BP*L*N*AD      = 1048576
  float* big   = y2buf + 1048576;          // 25165824 (aliased region)
  float* qkvb  = big;                      // BP*S*3AD       = 12582912
  float* attnb = big + 12582912;           // BP*S*AD        = 4194304
  float* upb   = big;                      // BP*S*2FH       = 16777216 (aliases qkv/attn)
  float* gateb = big + 16777216;           // BP*S*FH        = 8388608

  const long BLK = (long)O_ * D_;          // 65536: row-group stride in x_input / x

  // ---- phase A: norms + input projections -> x (interleaved old|new) ----
  rmsnorm_k<<<3072 / 8, 256, 0, stream>>>(x_input, D_, M_, BLK, 0, old_norm_w, xoldn);
  rmsnorm_k<<<1024 / 8, 256, 0, stream>>>(x_ar,    D_, 1024, 0, 0, new_norm_w, xarn);
  rmsnorm_k<<<1024 / 8, 256, 0, stream>>>(x_input, D_, N_, BLK, M_, new_norm_w, xden);

  launch_gemm(stream, xoldn, D_, 3072, 0, 0, w_old, xbuf, AD_, M_, BLK, 0,
              3072, AD_, D_, 0);
  launch_gemm(stream, xarn, D_, 1024, 0, 0, w_ar, xbuf, AD_, N_, BLK, M_,
              1024, AD_, D_, 0);
  launch_gemm(stream, xden, D_, 1024, 0, 0, w_de, xbuf, AD_, N_, BLK, M_,
              1024, AD_, D_, 1);

  const int NROWS = BP_ * S_;  // 4096
  for (int it = 0; it < 2; ++it) {
    for (int li = 0; li < 2; ++li) {
      // attention block
      rmsnorm_k<<<NROWS / 8, 256, 0, stream>>>(xbuf, AD_, NROWS, 0, 0,
                                               attn_norm_w + li * AD_, hbuf);
      launch_gemm(stream, hbuf, AD_, NROWS, 0, 0, w_qkv + (size_t)li * 3 * AD_ * AD_,
                  qkvb, 3 * AD_, NROWS, 0, 0, NROWS, 3 * AD_, AD_, 0);
      qk_prep<<<8192, 256, 0, stream>>>(qkvb);
      attn_flash<<<dim3(S_ / 64, AAN_, BP_), 128, 0, stream>>>(qkvb, doc, attnb);
      launch_gemm(stream, attnb, AD_, NROWS, 0, 0, w_o + (size_t)li * AD_ * AD_,
                  xbuf, AD_, NROWS, 0, 0, NROWS, AD_, AD_, 1);
      // FFN block
      rmsnorm_k<<<NROWS / 8, 256, 0, stream>>>(xbuf, AD_, NROWS, 0, 0,
                                               ffn_norm_w + li * AD_, hbuf);
      launch_gemm(stream, hbuf, AD_, NROWS, 0, 0, w_up + (size_t)li * 2 * FH_ * AD_,
                  upb, 2 * FH_, NROWS, 0, 0, NROWS, 2 * FH_, AD_, 0);
      silu_gate<<<(NROWS * FH_) / 256, 256, 0, stream>>>(upb, gateb);
      launch_gemm(stream, gateb, FH_, NROWS, 0, 0, w_down + (size_t)li * AD_ * FH_,
                  xbuf, AD_, NROWS, 0, 0, NROWS, AD_, FH_, 1);
    }
    if (it + 1 < 2) {
      // y2 from current x (must run before x is rebuilt)
      build_y2<<<128, 256, 0, stream>>>(xbuf, begin, pad_emb, new_norm_w, y2buf);
      // rebuild old columns and new columns of x for the next iteration
      launch_gemm(stream, xoldn, D_, 3072, 0, 0, w_old, xbuf, AD_, M_, BLK, 0,
                  3072, AD_, D_, 0);
      launch_gemm(stream, xarn, D_, 1024, 0, 0, w_ar1, xbuf, AD_, N_, BLK, M_,
                  1024, AD_, D_, 0);
      launch_gemm(stream, y2buf, AD_, 1024, 0, 0, w_y1, xbuf, AD_, N_, BLK, M_,
                  1024, AD_, AD_, 1);
      launch_gemm(stream, xden, D_, 1024, 0, 0, w_de1, xbuf, AD_, N_, BLK, M_,
                  1024, AD_, D_, 1);
    }
  }

  // final: out = y @ w_agg^T, y rows gathered from the "new" columns of x
  launch_gemm(stream, xbuf, AD_, N_, BLK, M_, w_agg, (float*)d_out, D_,
              1024, 0, 0, 1024, D_, AD_, 0);
}
